// OutputsTreeEmbedding_38886633898792
// MI455X (gfx1250) — matrix operand order repaired and checked by reading, not compiled, not run on hardware
//
#include <hip/hip_runtime.h>

// ---------------- problem constants ----------------
#define V_SIZE 100000
#define D_EMB  128
#define NL2    5000
#define NL3    500
#define NL4    50
#define B_SZ   64
#define S_SZ   512
#define NTOK   (B_SZ * S_SZ)   // 32768

typedef __bf16 bf16;
typedef __attribute__((ext_vector_type(16))) __bf16 v16bf;
typedef __attribute__((ext_vector_type(8)))  __bf16 v8bf;
typedef __attribute__((ext_vector_type(4)))  __bf16 v4bf;
typedef __attribute__((ext_vector_type(8)))  float  v8f;
typedef __attribute__((ext_vector_type(4)))  float  v4f;

// ---------------- workspace layout (float indices) ----------------
#define WS_L2    0
#define WS_L3    640000
#define WS_L4    704000
#define WS_ROOT  710400
#define WS_C2    710528
#define WS_C3    715528
#define WS_C4    716028
#define WS_NZERO 716078
#define WS_BF    716080
#define WS_BB    716592
#define WS_WPACK 717104   // bf16 region: 4 matrices of 512*128

// ---------------- phase 1: tree aggregation ----------------
__global__ void zero_kernel(float* p, int n) {
    int i = blockIdx.x * 256 + threadIdx.x;
    if (i < n) p[i] = 0.0f;
}

__global__ void scatter_kernel(const float* __restrict__ src,
                               const int* __restrict__ parent,
                               float* __restrict__ sums,
                               float* __restrict__ cnt,
                               int first, int nrows) {
    int idx = blockIdx.x * 256 + threadIdx.x;
    if (idx >= nrows * 128) return;
    int r = idx >> 7;
    int d = idx & 127;
    int v = first + r;
    int pa = parent[v];
    atomicAdd(&sums[(size_t)pa * 128 + d], src[(size_t)v * 128 + d]);
    if (d == 0) atomicAdd(&cnt[pa], 1.0f);
}

__global__ void norm_kernel(float* __restrict__ sums, const float* __restrict__ cnt, int nrows) {
    int idx = blockIdx.x * 256 + threadIdx.x;
    if (idx >= nrows * 128) return;
    float c = cnt[idx >> 7];
    c = c > 1.0f ? c : 1.0f;
    sums[idx] = sums[idx] / c;
}

__global__ void root_kernel(const float* __restrict__ L4, float* __restrict__ root) {
    int d = threadIdx.x;   // 128 threads
    float s = 0.0f;
    #pragma unroll 5
    for (int r = 0; r < NL4; ++r) s += L4[r * 128 + d];
    root[d] = s * (1.0f / (float)NL4);
}

__global__ void pack_kernel(const float* __restrict__ wihf, const float* __restrict__ whhf,
                            const float* __restrict__ wihb, const float* __restrict__ whhb,
                            const float* __restrict__ bihf, const float* __restrict__ bhhf,
                            const float* __restrict__ bihb, const float* __restrict__ bhhb,
                            bf16* __restrict__ wout,
                            float* __restrict__ bias_f, float* __restrict__ bias_b) {
    int idx = blockIdx.x * 256 + threadIdx.x;
    if (idx < 4 * 512 * 128) {
        int mat = idx >> 16;
        int off = idx & 65535;
        const float* src = (mat == 0) ? wihf : (mat == 1) ? whhf : (mat == 2) ? wihb : whhb;
        wout[idx] = (bf16)src[off];
    }
    if (idx < 512) bias_f[idx] = bihf[idx] + bhhf[idx];
    else if (idx < 1024) { int j = idx - 512; bias_b[j] = bihb[j] + bhhb[j]; }
}

// ---------------- WMMA fragment loaders ----------------
// A (16x32 bf16, M x K): lane L holds row M=L%16; halves K = (L/16)*8 .. +7 and +16..+23
__device__ __forceinline__ v16bf load_a_frag(const bf16* __restrict__ buf, int kbase, int lane) {
    int m  = lane & 15;
    int kh = (lane >> 4) * 8;
    const bf16* p = buf + m * 128 + kbase + kh;
    union { v16bf v; v8bf h[2]; } u;
    u.h[0] = *(const v8bf*)(p);
    u.h[1] = *(const v8bf*)(p + 16);
    return u.v;
}
// B (32x16 bf16, K x N): lane L holds col N=L%16; halves K = (L/16)*8 .. +7 and +16..+23
__device__ __forceinline__ v16bf load_b_frag(const bf16* __restrict__ rowp, int kbase, int khalf) {
    const bf16* p = rowp + kbase + khalf * 8;
    union { v16bf v; v8bf h[2]; } u;
    u.h[0] = *(const v8bf*)(p);
    u.h[1] = *(const v8bf*)(p + 16);
    return u.v;
}

__device__ __forceinline__ float fsig(float x) { return 1.0f / (1.0f + __expf(-x)); }

// ---------------- phase 2: bidirectional LSTM ----------------
// One wave (32 threads) per block; each wave owns 32 tokens = two 16-row
// M-tiles that SHARE every B fragment (1 B load feeds 2 WMMAs).
// Single-wave workgroup => all __syncthreads become S_NOP (compiler fence only).
__global__ __launch_bounds__(32)
void bilstm_kernel(const int* __restrict__ ids, const float* __restrict__ embed,
                   const int* __restrict__ p2, const int* __restrict__ p3, const int* __restrict__ p4,
                   const float* __restrict__ L2, const float* __restrict__ L3,
                   const float* __restrict__ L4, const float* __restrict__ root,
                   const bf16* __restrict__ Wf_ih, const bf16* __restrict__ Wf_hh,
                   const bf16* __restrict__ Wb_ih, const bf16* __restrict__ Wb_hh,
                   const float* __restrict__ bias_f, const float* __restrict__ bias_b,
                   float* __restrict__ out) {
    __shared__ bf16  xb[32 * 128];   // x_t staging, bf16   (8 KB)
    __shared__ bf16  hb[32 * 128];   // hidden state, bf16  (8 KB)
    __shared__ float cb[32 * 128];   // cell state, f32     (16 KB)
    __shared__ const float* ptrs[32][5];

    const int lane = threadIdx.x;    // 32 threads = 1 wave
    const int tokbase = blockIdx.x * 32;
    const int mrow  = lane & 15;
    const int khalf = lane >> 4;

    // Per-token source rows for the 5 steps (pad -> null, bos/eos -> leaf x5)
    {
        int tok = tokbase + lane;
        int id = ids[tok];
        const float *s0, *s1, *s2, *s3, *s4;
        if (id == 0) {
            s0 = s1 = s2 = s3 = s4 = nullptr;
        } else if (id == 1 || id == 2) {
            s0 = s1 = s2 = s3 = s4 = embed + (size_t)id * 128;
        } else {
            int a2 = p2[id];
            int a3 = p3[a2];
            int a4 = p4[a3];
            s0 = embed + (size_t)id * 128;
            s1 = L2 + (size_t)a2 * 128;
            s2 = L3 + (size_t)a3 * 128;
            s3 = L4 + (size_t)a4 * 128;
            s4 = root;
        }
        ptrs[lane][0] = s0; ptrs[lane][1] = s1; ptrs[lane][2] = s2;
        ptrs[lane][3] = s3; ptrs[lane][4] = s4;
    }
    __syncthreads();

    #pragma unroll 1
    for (int dir = 0; dir < 2; ++dir) {
        const bf16*  Wih  = dir ? Wb_ih  : Wf_ih;
        const bf16*  Whh  = dir ? Wb_hh  : Wf_hh;
        const float* bias = dir ? bias_b : bias_f;

        // zero h (bf16) and c (f32)
        {
            v4bf z;
            #pragma unroll
            for (int e = 0; e < 4; ++e) z[e] = (bf16)0.0f;
            for (int j = lane; j < 1024; j += 32) ((v4bf*)hb)[j] = z;
            v4f zf = {0.0f, 0.0f, 0.0f, 0.0f};
            for (int j = lane; j < 1024; j += 32) ((v4f*)cb)[j] = zf;
        }
        __syncthreads();

        #pragma unroll 1
        for (int s = 0; s < 5; ++s) {
            const int t = dir ? (4 - s) : s;

            // ---- stage x_t: each lane converts one 128-wide token row ----
            {
                const float* src = ptrs[lane][t];
                bf16* drow = xb + lane * 128;
                if (src) {
                    const v4f* s4p = (const v4f*)src;
                    #pragma unroll
                    for (int j = 0; j < 32; ++j) {
                        v4f v = s4p[j];
                        v4bf o;
                        o[0] = (bf16)v[0]; o[1] = (bf16)v[1];
                        o[2] = (bf16)v[2]; o[3] = (bf16)v[3];
                        ((v4bf*)drow)[j] = o;
                    }
                } else {
                    v4bf z;
                    #pragma unroll
                    for (int e = 0; e < 4; ++e) z[e] = (bf16)0.0f;
                    #pragma unroll
                    for (int j = 0; j < 32; ++j) ((v4bf*)drow)[j] = z;
                }
            }
            __syncthreads();

            const bool last = (s == 4);
            const int  rowoff = khalf * 8;

            #pragma unroll 1
            for (int nt = 0; nt < 8; ++nt) {
                const int nb = nt * 16;

                // bias-splat accumulators: [m-tile][gate], gates i,f,g,o
                v8f acc[2][4];
                #pragma unroll
                for (int g = 0; g < 4; ++g) {
                    float bv = bias[g * 128 + nb + mrow];
                    #pragma unroll
                    for (int e = 0; e < 8; ++e) acc[0][g][e] = bv;
                    acc[1][g] = acc[0][g];
                }

                // x @ W_ih^T : A frags re-read from LDS per k, B shared by 2 tiles
                #pragma unroll
                for (int k = 0; k < 4; ++k) {
                    v16bf a0 = load_a_frag(xb,            k * 32, lane);
                    v16bf a1 = load_a_frag(xb + 16 * 128, k * 32, lane);
                    #pragma unroll
                    for (int g = 0; g < 4; ++g) {
                        v16bf b = load_b_frag(Wih + (size_t)(g * 128 + nb + mrow) * 128, k * 32, khalf);
                        acc[0][g] = __builtin_amdgcn_wmma_f32_16x16x32_bf16(
                            false, a0, false, b, (short)0, acc[0][g], false, false);
                        acc[1][g] = __builtin_amdgcn_wmma_f32_16x16x32_bf16(
                            false, a1, false, b, (short)0, acc[1][g], false, false);
                    }
                }
                // h @ W_hh^T
                #pragma unroll
                for (int k = 0; k < 4; ++k) {
                    v16bf a0 = load_a_frag(hb,            k * 32, lane);
                    v16bf a1 = load_a_frag(hb + 16 * 128, k * 32, lane);
                    #pragma unroll
                    for (int g = 0; g < 4; ++g) {
                        v16bf b = load_b_frag(Whh + (size_t)(g * 128 + nb + mrow) * 128, k * 32, khalf);
                        acc[0][g] = __builtin_amdgcn_wmma_f32_16x16x32_bf16(
                            false, a0, false, b, (short)0, acc[0][g], false, false);
                        acc[1][g] = __builtin_amdgcn_wmma_f32_16x16x32_bf16(
                            false, a1, false, b, (short)0, acc[1][g], false, false);
                    }
                }

                // ---- gate nonlinearity + cell update (c in LDS, f32) ----
                #pragma unroll
                for (int tile = 0; tile < 2; ++tile) {
                    #pragma unroll
                    for (int e = 0; e < 8; ++e) {
                        int row = tile * 16 + e + rowoff;
                        int off = row * 128 + nb + mrow;
                        float iv = fsig(acc[tile][0][e]);
                        float fv = fsig(acc[tile][1][e]);
                        float gv = tanhf(acc[tile][2][e]);
                        float ov = fsig(acc[tile][3][e]);
                        float cv = fv * cb[off] + iv * gv;
                        cb[off] = cv;
                        float hv = ov * tanhf(cv);
                        hb[off] = (bf16)hv;
                        if (last)
                            out[(size_t)(tokbase + row) * 256 + dir * 128 + nb + mrow] = hv;
                    }
                }
            }
            __syncthreads();
        }
        __syncthreads();
    }
}

// ---------------- launcher ----------------
extern "C" void kernel_launch(void* const* d_in, const int* in_sizes, int n_in,
                              void* d_out, int out_size, void* d_ws, size_t ws_size,
                              hipStream_t stream) {
    const int*   input_src = (const int*)  d_in[0];
    const float* embed     = (const float*)d_in[1];
    const float* w_ih_f    = (const float*)d_in[2];
    const float* w_hh_f    = (const float*)d_in[3];
    const float* b_ih_f    = (const float*)d_in[4];
    const float* b_hh_f    = (const float*)d_in[5];
    const float* w_ih_b    = (const float*)d_in[6];
    const float* w_hh_b    = (const float*)d_in[7];
    const float* b_ih_b    = (const float*)d_in[8];
    const float* b_hh_b    = (const float*)d_in[9];
    const int*   parent_l2 = (const int*)  d_in[10];
    const int*   parent_l3 = (const int*)  d_in[11];
    const int*   parent_l4 = (const int*)  d_in[12];
    float* out = (float*)d_out;

    float* F      = (float*)d_ws;
    float* L2s    = F + WS_L2;
    float* L3s    = F + WS_L3;
    float* L4s    = F + WS_L4;
    float* rootp  = F + WS_ROOT;
    float* c2     = F + WS_C2;
    float* c3     = F + WS_C3;
    float* c4     = F + WS_C4;
    float* bias_f = F + WS_BF;
    float* bias_b = F + WS_BB;
    bf16*  wpack  = (bf16*)(F + WS_WPACK);
    bf16*  Wf_ih  = wpack;
    bf16*  Wf_hh  = wpack + 1 * 512 * 128;
    bf16*  Wb_ih  = wpack + 2 * 512 * 128;
    bf16*  Wb_hh  = wpack + 3 * 512 * 128;

    zero_kernel<<<(WS_NZERO + 255) / 256, 256, 0, stream>>>(F, WS_NZERO);

    pack_kernel<<<(4 * 512 * 128 + 255) / 256, 256, 0, stream>>>(
        w_ih_f, w_hh_f, w_ih_b, w_hh_b, b_ih_f, b_hh_f, b_ih_b, b_hh_b,
        wpack, bias_f, bias_b);

    scatter_kernel<<<((V_SIZE - 3) * 128 + 255) / 256, 256, 0, stream>>>(
        embed, parent_l2, L2s, c2, 3, V_SIZE - 3);
    norm_kernel<<<(NL2 * 128 + 255) / 256, 256, 0, stream>>>(L2s, c2, NL2);
    scatter_kernel<<<(NL2 * 128 + 255) / 256, 256, 0, stream>>>(
        L2s, parent_l3, L3s, c3, 0, NL2);
    norm_kernel<<<(NL3 * 128 + 255) / 256, 256, 0, stream>>>(L3s, c3, NL3);
    scatter_kernel<<<(NL3 * 128 + 255) / 256, 256, 0, stream>>>(
        L3s, parent_l4, L4s, c4, 0, NL3);
    norm_kernel<<<(NL4 * 128 + 255) / 256, 256, 0, stream>>>(L4s, c4, NL4);
    root_kernel<<<1, 128, 0, stream>>>(L4s, rootp);

    // bidirectional LSTM: 1024 blocks x 32 threads (1 wave, 32 tokens each)
    bilstm_kernel<<<NTOK / 32, 32, 0, stream>>>(
        input_src, embed, parent_l2, parent_l3, parent_l4,
        L2s, L3s, L4s, rootp,
        Wf_ih, Wf_hh, Wb_ih, Wb_hh, bias_f, bias_b, out);
}
